// MedformerLayer_38843684225392
// MI455X (gfx1250) — matrix operand order repaired and checked by reading, not compiled, not run on hardware
//
#include <hip/hip_runtime.h>
#include <hip/hip_bf16.h>

typedef _Float16 half16 __attribute__((ext_vector_type(16)));
typedef _Float16 half8  __attribute__((ext_vector_type(8)));
typedef float    float8 __attribute__((ext_vector_type(8)));

constexpr int kNB = 4, kB = 4, kL = 1024, kD = 512, kH = 8, kE = 64;
constexpr int kM = kNB * kB * kL;          // 16384 rows for the big GEMMs
constexpr float kScale = 0.125f;           // 1/sqrt(64)

// ---------------------------------------------------------------------------
// WMMA fragment helpers (CDNA5 16x16x32 f16 layouts, ISA 7.12.2)
// A (MxK=16x32): lanes 0-15 -> row=lane, K {0..7,16..23}; lanes 16-31 -> K {8..15,24..31}
// B (KxN=32x16): lanes 0-15 -> col=lane, K {0..7,16..23}; lanes 16-31 -> K {8..15,24..31}
// C/D (16x16 f32): VGPR r -> row r (lanes 0-15) / row r+8 (lanes 16-31), lane -> col
// ---------------------------------------------------------------------------
__device__ __forceinline__ half16 afrag_load(const _Float16* base, int ld,
                                             int m0, int k0, int lane) {
  const int row = m0 + (lane & 15);
  const int kb  = k0 + ((lane & 16) ? 8 : 0);
  const _Float16* p = base + (long)row * ld + kb;
  half16 a;
  ((half8*)&a)[0] = *(const half8*)(p);
  ((half8*)&a)[1] = *(const half8*)(p + 16);
  return a;
}

__device__ __forceinline__ half16 bfrag_load(const _Float16* base, int ld,
                                             int k0, int n0, int lane) {
  const _Float16* p = base + (long)(k0 + ((lane & 16) ? 8 : 0)) * ld + n0 + (lane & 15);
  half16 b;
#pragma unroll
  for (int s = 0; s < 16; ++s)
    b[s] = p[(long)((s & 7) + ((s & 8) ? 16 : 0)) * ld];
  return b;
}

__device__ __forceinline__ float8 wmma32(half16 a, half16 b, float8 c) {
  return __builtin_amdgcn_wmma_f32_16x16x32_f16(false, a, false, b,
                                                (short)0, c, false, false);
}

// ---------------------------------------------------------------------------
// f32 -> f16 conversion
// ---------------------------------------------------------------------------
__global__ void cvt_f16_kernel(const float* __restrict__ s,
                               _Float16* __restrict__ d, int n) {
  int i = blockIdx.x * blockDim.x + threadIdx.x;
  int stride = gridDim.x * blockDim.x;
  for (; i < n; i += stride) d[i] = (_Float16)s[i];
}

// ---------------------------------------------------------------------------
// QKV projection GEMM: [16384 x 512] @ [512 x 512] per block/projection.
// blockIdx.z selects projection (0=q scaled, 1=k transposed, 2=v).
// Wave = 32x64 output tile; 8 waves per block -> 64x256 per block.
// ---------------------------------------------------------------------------
__global__ void __launch_bounds__(256)
qkv_gemm_kernel(const _Float16* __restrict__ xh,
                const _Float16* __restrict__ wq,
                const _Float16* __restrict__ wk,
                const _Float16* __restrict__ wv,
                const float* __restrict__ bq,
                const float* __restrict__ bk,
                const float* __restrict__ bv,
                _Float16* __restrict__ qo,    // [nb,b,h, l, e]  (pre-scaled)
                _Float16* __restrict__ kto,   // [nb,b,h, e, l]  (transposed)
                _Float16* __restrict__ vo) {  // [nb,b,h, l, e]
  const int lane = threadIdx.x & 31;
  const int wid  = threadIdx.x >> 5;
  const int m0 = blockIdx.x * 64 + (wid >> 2) * 32;
  const int n0 = blockIdx.y * 256 + (wid & 3) * 64;
  const int proj = blockIdx.z;
  const int nblk = m0 / (kB * kL);

  const _Float16* W = (proj == 0 ? wq : proj == 1 ? wk : wv) + (long)nblk * kD * kD;
  const float* bias = (proj == 0 ? bq : proj == 1 ? bk : bv) + nblk * kD;

  float8 acc[2][4] = {};
  for (int k0 = 0; k0 < kD; k0 += 32) {
    half16 a0 = afrag_load(xh, kD, m0,      k0, lane);
    half16 a1 = afrag_load(xh, kD, m0 + 16, k0, lane);
#pragma unroll
    for (int nt = 0; nt < 4; ++nt) {
      half16 bf = bfrag_load(W, kD, k0, n0 + nt * 16, lane);
      acc[0][nt] = wmma32(a0, bf, acc[0][nt]);
      acc[1][nt] = wmma32(a1, bf, acc[1][nt]);
    }
  }

  const int col = lane & 15;
  const int hio = (lane & 16) ? 8 : 0;
#pragma unroll
  for (int mt = 0; mt < 2; ++mt) {
#pragma unroll
    for (int nt = 0; nt < 4; ++nt) {
      const int dcol = n0 + nt * 16 + col;
      const float bval = bias[dcol];
      const int h = dcol >> 6, e = dcol & 63;
#pragma unroll
      for (int j = 0; j < 8; ++j) {
        const int m = m0 + mt * 16 + hio + j;
        const int b = (m >> 10) & 3, l = m & 1023;
        const int hh = ((nblk * kB + b) * kH + h);
        const float vf = acc[mt][nt][j] + bval;
        if (proj == 0)      qo[(long)hh * kL * kE + l * kE + e] = (_Float16)(vf * kScale);
        else if (proj == 1) kto[(long)hh * kE * kL + e * kL + l] = (_Float16)vf;
        else                vo[(long)hh * kL * kE + l * kE + e] = (_Float16)vf;
      }
    }
  }
}

// ---------------------------------------------------------------------------
// Flash attention per head. 128 heads, 8 blocks/head, 8 waves/block,
// each wave owns 16 query rows and streams 64-wide K/V tiles.
// ---------------------------------------------------------------------------
__global__ void __launch_bounds__(256)
attn_kernel(const _Float16* __restrict__ qh,
            const _Float16* __restrict__ kth,
            const _Float16* __restrict__ vh,
            _Float16* __restrict__ oh) {   // [nb,b,l,d]
  __shared__ _Float16 plds[8 * 16 * 64];   // 2KB per wave

  const int lane = threadIdx.x & 31;
  const int wid  = threadIdx.x >> 5;
  const int hh    = blockIdx.x >> 3;       // (nb*B + b)*H + h
  const int chunk = blockIdx.x & 7;
  const int l0 = chunk * 128 + wid * 16;

  const _Float16* qb  = qh  + (long)hh * kL * kE;
  const _Float16* ktb = kth + (long)hh * kE * kL;
  const _Float16* vb  = vh  + (long)hh * kL * kE;
  _Float16* pw = plds + wid * 1024;

  half16 aq0 = afrag_load(qb, kE, l0, 0,  lane);
  half16 aq1 = afrag_load(qb, kE, l0, 32, lane);

  float8 acc[4] = {};
  float m[8], ls[8];
#pragma unroll
  for (int r = 0; r < 8; ++r) { m[r] = -1e30f; ls[r] = 0.f; }

  const int col = lane & 15;
  const int hio = (lane & 16) ? 8 : 0;

  for (int s0 = 0; s0 < kL; s0 += 64) {
    // ---- scores: S[16 x 64] = Q[16x64] @ K^T[64 x 64-cols] ----
    float8 sc[4] = {};
#pragma unroll
    for (int nt = 0; nt < 4; ++nt) {
      half16 bk0 = bfrag_load(ktb, kL, 0,  s0 + nt * 16, lane);
      half16 bk1 = bfrag_load(ktb, kL, 32, s0 + nt * 16, lane);
      sc[nt] = wmma32(aq0, bk0, sc[nt]);
      sc[nt] = wmma32(aq1, bk1, sc[nt]);
    }
    // ---- online softmax (row stats across 16-lane groups) ----
#pragma unroll
    for (int r = 0; r < 8; ++r) {
      float mt = sc[0][r];
      mt = fmaxf(mt, sc[1][r]); mt = fmaxf(mt, sc[2][r]); mt = fmaxf(mt, sc[3][r]);
      mt = fmaxf(mt, __shfl_xor(mt, 1, 32));
      mt = fmaxf(mt, __shfl_xor(mt, 2, 32));
      mt = fmaxf(mt, __shfl_xor(mt, 4, 32));
      mt = fmaxf(mt, __shfl_xor(mt, 8, 32));
      const float mn = fmaxf(m[r], mt);
      const float alpha = __expf(m[r] - mn);
      m[r] = mn;
      float rs = 0.f;
#pragma unroll
      for (int nt = 0; nt < 4; ++nt) {
        const float p = __expf(sc[nt][r] - mn);
        sc[nt][r] = p;
        rs += p;
      }
      rs += __shfl_xor(rs, 1, 32);
      rs += __shfl_xor(rs, 2, 32);
      rs += __shfl_xor(rs, 4, 32);
      rs += __shfl_xor(rs, 8, 32);
      ls[r] = ls[r] * alpha + rs;
#pragma unroll
      for (int f = 0; f < 4; ++f) acc[f][r] *= alpha;
    }
    // ---- P to LDS (C-layout -> row-major f16) ----
#pragma unroll
    for (int nt = 0; nt < 4; ++nt)
#pragma unroll
      for (int r = 0; r < 8; ++r)
        pw[(r + hio) * 64 + nt * 16 + col] = (_Float16)sc[nt][r];
    __asm__ volatile("" ::: "memory");
    // ---- O += P[16x64] @ V[64 x 64] ----
#pragma unroll
    for (int c = 0; c < 2; ++c) {
      half16 ap = afrag_load(pw, 64, 0, c * 32, lane);
#pragma unroll
      for (int nt = 0; nt < 4; ++nt) {
        half16 bv = bfrag_load(vb, kE, s0 + c * 32, nt * 16, lane);
        acc[nt] = wmma32(ap, bv, acc[nt]);
      }
    }
  }

  // ---- normalize + store O as [nb,b,l,d] f16 ----
  const int nb_b = hh >> 3;     // nb*B + b
  const int h    = hh & 7;
#pragma unroll
  for (int f = 0; f < 4; ++f)
#pragma unroll
    for (int r = 0; r < 8; ++r) {
      const int l = l0 + hio + r;
      const long mrow = (long)nb_b * kL + l;
      oh[mrow * kD + h * kE + f * 16 + col] = (_Float16)(acc[f][r] / ls[r]);
    }
}

// ---------------------------------------------------------------------------
// Output projection GEMM: x_intra = O @ Wo + bo -> f32 d_out
// ---------------------------------------------------------------------------
__global__ void __launch_bounds__(256)
oproj_gemm_kernel(const _Float16* __restrict__ ohh,
                  const _Float16* __restrict__ woh,
                  const float* __restrict__ bo,
                  float* __restrict__ out) {
  const int lane = threadIdx.x & 31;
  const int wid  = threadIdx.x >> 5;
  const int m0 = blockIdx.x * 64 + (wid >> 2) * 32;
  const int n0 = blockIdx.y * 256 + (wid & 3) * 64;
  const int nblk = m0 / (kB * kL);

  const _Float16* W = woh + (long)nblk * kD * kD;
  const float* bias = bo + nblk * kD;

  float8 acc[2][4] = {};
  for (int k0 = 0; k0 < kD; k0 += 32) {
    half16 a0 = afrag_load(ohh, kD, m0,      k0, lane);
    half16 a1 = afrag_load(ohh, kD, m0 + 16, k0, lane);
#pragma unroll
    for (int nt = 0; nt < 4; ++nt) {
      half16 bf = bfrag_load(W, kD, k0, n0 + nt * 16, lane);
      acc[0][nt] = wmma32(a0, bf, acc[0][nt]);
      acc[1][nt] = wmma32(a1, bf, acc[1][nt]);
    }
  }

  const int col = lane & 15;
  const int hio = (lane & 16) ? 8 : 0;
#pragma unroll
  for (int mt = 0; mt < 2; ++mt)
#pragma unroll
    for (int nt = 0; nt < 4; ++nt) {
      const int dcol = n0 + nt * 16 + col;
      const float bval = bias[dcol];
#pragma unroll
      for (int j = 0; j < 8; ++j) {
        const int mrow = m0 + mt * 16 + hio + j;
        out[(long)mrow * kD + dcol] = acc[mt][nt][j] + bval;
      }
    }
}

// ---------------------------------------------------------------------------
// Router (inter) attention: tiny, scalar. 16 tokens total.
// ---------------------------------------------------------------------------
__global__ void __launch_bounds__(256)
inter_proj_kernel(const float* __restrict__ out,   // d_out (has x_intra)
                  const float* __restrict__ iWq, const float* __restrict__ ibq,
                  const float* __restrict__ iWk, const float* __restrict__ ibk,
                  const float* __restrict__ iWv, const float* __restrict__ ibv,
                  float* __restrict__ rq, float* __restrict__ rk,
                  float* __restrict__ rv) {
  __shared__ float r[kD];
  const int t = blockIdx.x;           // token: b*NB + nb
  const int b = t >> 2, nb = t & 3;
  const int proj = blockIdx.y;
  const float* src = out + ((long)(nb * kB + b) * kL + (kL - 1)) * kD;
  for (int i = threadIdx.x; i < kD; i += 256) r[i] = src[i];
  __syncthreads();

  const float* W = proj == 0 ? iWq : proj == 1 ? iWk : iWv;
  const float* bb = proj == 0 ? ibq : proj == 1 ? ibk : ibv;
  float* dst = proj == 0 ? rq : proj == 1 ? rk : rv;

  for (int d = threadIdx.x; d < kD; d += 256) {
    float acc = bb[d];
    for (int k = 0; k < kD; ++k) acc += r[k] * W[k * kD + d];
    if (proj == 0) acc *= kScale;
    dst[t * kD + d] = acc;
  }
}

__global__ void __launch_bounds__(128)
inter_attn_kernel(const float* __restrict__ rq, const float* __restrict__ rk,
                  const float* __restrict__ rv, float* __restrict__ ro) {
  const int t = threadIdx.x;          // b(2) | nq(2) | h(3)
  const int b = t >> 5, nq = (t >> 3) & 3, h = t & 7;
  const float* qr = rq + (b * kNB + nq) * kD + h * kE;
  float sc[kNB], mx = -1e30f;
  for (int nk = 0; nk < kNB; ++nk) {
    const float* kr = rk + (b * kNB + nk) * kD + h * kE;
    float s = 0.f;
    for (int e = 0; e < kE; ++e) s += qr[e] * kr[e];
    sc[nk] = s;
    mx = fmaxf(mx, s);
  }
  float den = 0.f;
  for (int nk = 0; nk < kNB; ++nk) { sc[nk] = __expf(sc[nk] - mx); den += sc[nk]; }
  float* orow = ro + (b * kNB + nq) * kD + h * kE;
  for (int e = 0; e < kE; ++e) {
    float acc = 0.f;
    for (int nk = 0; nk < kNB; ++nk)
      acc += sc[nk] * rv[(b * kNB + nk) * kD + h * kE + e];
    orow[e] = acc / den;
  }
}

__global__ void __launch_bounds__(256)
inter_out_kernel(const float* __restrict__ ro, const float* __restrict__ iWo,
                 const float* __restrict__ ibo, float* __restrict__ out) {
  __shared__ float r[kD];
  const int t = blockIdx.x;           // b*NB + nb
  const int b = t >> 2, nb = t & 3;
  for (int i = threadIdx.x; i < kD; i += 256) r[i] = ro[t * kD + i];
  __syncthreads();
  float* dst = out + ((long)(nb * kB + b) * kL + (kL - 1)) * kD;
  for (int d = threadIdx.x; d < kD; d += 256) {
    float acc = ibo[d];
    for (int k = 0; k < kD; ++k) acc += r[k] * iWo[k * kD + d];
    dst[d] = acc;
  }
}

// ---------------------------------------------------------------------------
extern "C" void kernel_launch(void* const* d_in, const int* in_sizes, int n_in,
                              void* d_out, int out_size, void* d_ws, size_t ws_size,
                              hipStream_t stream) {
  const float* x   = (const float*)d_in[0];
  const float* Wq  = (const float*)d_in[1];  const float* bq  = (const float*)d_in[2];
  const float* Wk  = (const float*)d_in[3];  const float* bk  = (const float*)d_in[4];
  const float* Wv  = (const float*)d_in[5];  const float* bv  = (const float*)d_in[6];
  const float* Wo  = (const float*)d_in[7];  const float* bo  = (const float*)d_in[8];
  const float* iWq = (const float*)d_in[9];  const float* ibq = (const float*)d_in[10];
  const float* iWk = (const float*)d_in[11]; const float* ibk = (const float*)d_in[12];
  const float* iWv = (const float*)d_in[13]; const float* ibv = (const float*)d_in[14];
  const float* iWo = (const float*)d_in[15]; const float* ibo = (const float*)d_in[16];
  float* out = (float*)d_out;

  const long nX = (long)kM * kD;        // 8,388,608
  const long nW = (long)kNB * kD * kD;  // 1,048,576

  char* ws = (char*)d_ws;
  size_t off = 0;
  _Float16* xh  = (_Float16*)(ws + off); off += nX * 2;
  _Float16* wqh = (_Float16*)(ws + off); off += nW * 2;
  _Float16* wkh = (_Float16*)(ws + off); off += nW * 2;
  _Float16* wvh = (_Float16*)(ws + off); off += nW * 2;
  _Float16* woh = (_Float16*)(ws + off); off += nW * 2;
  _Float16* qh  = (_Float16*)(ws + off); off += nX * 2;
  _Float16* kth = (_Float16*)(ws + off); off += nX * 2;
  _Float16* vh  = (_Float16*)(ws + off); off += nX * 2;
  _Float16* oh  = (_Float16*)(ws + off); off += nX * 2;
  float* rq = (float*)(ws + off); off += 16 * kD * 4;
  float* rk = (float*)(ws + off); off += 16 * kD * 4;
  float* rv = (float*)(ws + off); off += 16 * kD * 4;
  float* ro = (float*)(ws + off); off += 16 * kD * 4;

  // 1) f32 -> f16 converts
  cvt_f16_kernel<<<4096, 256, 0, stream>>>(x,  xh,  (int)nX);
  cvt_f16_kernel<<<1024, 256, 0, stream>>>(Wq, wqh, (int)nW);
  cvt_f16_kernel<<<1024, 256, 0, stream>>>(Wk, wkh, (int)nW);
  cvt_f16_kernel<<<1024, 256, 0, stream>>>(Wv, wvh, (int)nW);
  cvt_f16_kernel<<<1024, 256, 0, stream>>>(Wo, woh, (int)nW);

  // 2) QKV projections (WMMA)
  qkv_gemm_kernel<<<dim3(kM / 64, kD / 256, 3), 256, 0, stream>>>(
      xh, wqh, wkh, wvh, bq, bk, bv, qh, kth, vh);

  // 3) flash attention per head (WMMA)
  attn_kernel<<<dim3(kNB * kB * kH * 8), 256, 0, stream>>>(qh, kth, vh, oh);

  // 4) output projection (WMMA) -> f32 d_out
  oproj_gemm_kernel<<<dim3(kM / 64, kD / 256), 256, 0, stream>>>(oh, woh, bo, out);

  // 5) router (inter) attention, overwrites the 16 router rows of d_out
  inter_proj_kernel<<<dim3(16, 3), 256, 0, stream>>>(
      out, iWq, ibq, iWk, ibk, iWv, ibv, rq, rk, rv);
  inter_attn_kernel<<<1, 128, 0, stream>>>(rq, rk, rv, ro);
  inter_out_kernel<<<16, 256, 0, stream>>>(ro, iWo, ibo, out);
}
